// Transformer_31009663877899
// MI455X (gfx1250) — matrix operand order repaired and checked by reading, not compiled, not run on hardware
//
#include <hip/hip_runtime.h>
#include <hip/hip_bf16.h>
#include <math.h>

// ---------------- types ----------------
typedef __bf16 bf16;
typedef __bf16 v16bf __attribute__((ext_vector_type(16)));
typedef __bf16 v8bf  __attribute__((ext_vector_type(8)));
typedef float  v8f   __attribute__((ext_vector_type(8)));
typedef unsigned u32x4 __attribute__((ext_vector_type(4)));
typedef unsigned u32x8 __attribute__((ext_vector_type(8)));

#define VCAB 32000
#define EDIM 768
#define HNUM 12
#define HDIM 64
#define LNUM 6
#define TSEQ 1024
#define BBAT 4
#define MTOK (BBAT * TSEQ)   // 4096

// ---------------- LDS address helper (32-bit DS offset) ----------------
template <typename T>
__device__ __forceinline__ unsigned lds_addr(T* p) {
    return (unsigned)(size_t)(__attribute__((address_space(3))) T*)p;
}

// ---------------- Tensor Data Mover: 2-D tile global -> LDS ----------------
// D# per CDNA5 ISA ch.8: group0 = {count/lds_addr/global_addr/type},
// group1 = {data_size, tensor_dim0/1, tile_dim0/1, tensor_dim0_stride}.
// 2-byte elements. Groups 2/3 zero-filled (tile_dim2/3 = 0 -> unused).
__device__ __forceinline__ void tdm_load_2d(unsigned lds_byte_addr, const void* gsrc,
                                            unsigned tensor_d0, unsigned tensor_d1,
                                            unsigned stride_d0,
                                            unsigned tile_d0, unsigned tile_d1) {
    unsigned long long ga = (unsigned long long)(size_t)gsrc;
    u32x4 g0;
    g0[0] = 1u;                                            // count=1, user mode
    g0[1] = lds_byte_addr;                                 // lds_addr
    g0[2] = (unsigned)(ga & 0xFFFFFFFFu);                  // global_addr[31:0]
    g0[3] = (unsigned)((ga >> 32) & 0x01FFFFFFu) | (2u << 30);  // ga[56:32] | type=2
    u32x8 g1;
    g1[0] = 1u << 16;                                      // data_size = 1 (2 bytes)
    g1[1] = (tensor_d0 & 0xFFFFu) << 16;                   // tensor_dim0[15:0] @bit48
    g1[2] = (tensor_d0 >> 16) | ((tensor_d1 & 0xFFFFu) << 16); // dim0 hi | dim1 lo
    g1[3] = (tensor_d1 >> 16) | ((tile_d0 & 0xFFFFu) << 16);   // dim1 hi | tile_dim0
    g1[4] = (tile_d1 & 0xFFFFu);                           // tile_dim1 | tile_dim2=0
    g1[5] = stride_d0;                                     // tensor_dim0_stride[31:0]
    g1[6] = 0u;                                            // stride hi | dim1_stride lo
    g1[7] = 0u;
    u32x4 gz = {0u, 0u, 0u, 0u};
    asm volatile("tensor_load_to_lds %0, %1, %2, %3"
                 :: "s"(g0), "s"(g1), "s"(gz), "s"(gz)
                 : "memory");
}

__device__ __forceinline__ void wait_ds0() {
    asm volatile("s_wait_dscnt 0x0" ::: "memory");
}

// ---------------- DPP16 butterfly reductions over 16-lane rows ----------------
template <int CTRL>
__device__ __forceinline__ float dpp_perm_f(float x) {
    int y = __builtin_amdgcn_update_dpp(0, __float_as_int(x), CTRL, 0xF, 0xF, true);
    return __int_as_float(y);
}
__device__ __forceinline__ float rmax16(float v) {
    v = fmaxf(v, dpp_perm_f<0xB1>(v));    // quad_perm [1,0,3,2]  (xor 1)
    v = fmaxf(v, dpp_perm_f<0x4E>(v));    // quad_perm [2,3,0,1]  (xor 2)
    v = fmaxf(v, dpp_perm_f<0x141>(v));   // row_half_mirror
    v = fmaxf(v, dpp_perm_f<0x140>(v));   // row_mirror
    return v;
}
__device__ __forceinline__ float rsum16(float v) {
    v += dpp_perm_f<0xB1>(v);
    v += dpp_perm_f<0x4E>(v);
    v += dpp_perm_f<0x141>(v);
    v += dpp_perm_f<0x140>(v);
    return v;
}

// =====================================================================
// fp32 -> bf16 conversion (weights, once per launch)
// =====================================================================
__global__ void f32_to_bf16_kernel(const float* __restrict__ in,
                                   bf16* __restrict__ out, int n) {
    int i = blockIdx.x * blockDim.x + threadIdx.x;
    int stride = gridDim.x * blockDim.x;
    for (; i < n; i += stride) out[i] = (bf16)in[i];
}

// =====================================================================
// Embedding
// =====================================================================
__global__ void embed_kernel(const int* __restrict__ idx,
                             const float* __restrict__ tok,
                             const float* __restrict__ pos,
                             float* __restrict__ x) {
    int row = blockIdx.x;             // b*T + t
    int t = row & (TSEQ - 1);
    int tk = idx[row];
    const float* tr = tok + (size_t)tk * EDIM;
    const float* pr = pos + (size_t)t * EDIM;
    float* xr = x + (size_t)row * EDIM;
    for (int i = threadIdx.x; i < EDIM; i += blockDim.x)
        xr[i] = tr[i] + pr[i];
}

// =====================================================================
// LayerNorm (fp32 in) -> bf16 out; one block per row
// =====================================================================
__global__ void layernorm_bf16_kernel(const float* __restrict__ x,
                                      const float* __restrict__ g,
                                      const float* __restrict__ b,
                                      bf16* __restrict__ out) {
    __shared__ float red[256];
    int row = blockIdx.x;
    const float* xr = x + (size_t)row * EDIM;
    float s = 0.f, s2 = 0.f;
    for (int i = threadIdx.x; i < EDIM; i += 256) {
        float v = xr[i];
        s += v; s2 += v * v;
    }
    red[threadIdx.x] = s;
    __syncthreads();
    for (int st = 128; st > 0; st >>= 1) {
        if (threadIdx.x < st) red[threadIdx.x] += red[threadIdx.x + st];
        __syncthreads();
    }
    float mean = red[0] * (1.0f / EDIM);
    __syncthreads();
    red[threadIdx.x] = s2;
    __syncthreads();
    for (int st = 128; st > 0; st >>= 1) {
        if (threadIdx.x < st) red[threadIdx.x] += red[threadIdx.x + st];
        __syncthreads();
    }
    float var = red[0] * (1.0f / EDIM) - mean * mean;
    float inv = rsqrtf(var + 1e-5f);
    bf16* orow = out + (size_t)row * EDIM;
    for (int i = threadIdx.x; i < EDIM; i += 256) {
        float v = (xr[i] - mean) * inv * g[i] + b[i];
        orow[i] = (bf16)v;
    }
}

// =====================================================================
// WMMA fragment loader for A (16x32 MxK), works for global or LDS base.
// lane<16: row=lane, halves 0..7 = K+0..7, 8..15 = K+16..23
// lane>=16: row=lane-16, K offsets +8 / +24
// =====================================================================
__device__ __forceinline__ v16bf load_a_frag(const bf16* __restrict__ base,
                                             int stride, int lane) {
    const bf16* p = base + (size_t)(lane & 15) * stride + ((lane & 16) ? 8 : 0);
    union { v16bf v; v8bf h[2]; } u;
    u.h[0] = *(const v8bf*)(p);
    u.h[1] = *(const v8bf*)(p + 16);
    return u.v;
}

// =====================================================================
// GEMM: C[M,N] = A[M,K](bf16) @ W[N,K]^T(bf16) + bias
//  MODE 0: fp32    1: bf16    2: fp32 + residual    3: bf16 gelu
// block = 256 thr (8 waves); block tile 256x64; wave tile 32x64.
// A (256x32) and B (64x32) tiles double-buffered in LDS, filled by the
// Tensor Data Mover (wave 0 issues descriptors; TENSORcnt pipelined).
// grid = (N/64, M/256)
// =====================================================================
template <int MODE>
__global__ void gemm_wmma_kernel(const bf16* __restrict__ A,
                                 const bf16* __restrict__ W,
                                 const float* __restrict__ bias,
                                 const float* __restrict__ res,
                                 float* __restrict__ outf,
                                 bf16* __restrict__ outb,
                                 int M, int N, int K) {
    __shared__ __align__(16) bf16 At[2][256 * 32];   // 32KB
    __shared__ __align__(16) bf16 Bt[2][64 * 32];    // 8KB

    int tid = threadIdx.x;
    int lane = tid & 31;
    int wave = tid >> 5;
    int blockRow = blockIdx.y * 256;
    int wrow = blockRow + wave * 32;
    int n0   = blockIdx.x * 64;

    const bf16* Ag = A + (size_t)blockRow * K;   // block A tile base (k=0)
    const bf16* Wg = W + (size_t)n0 * K;         // block B tile base (k=0)
    bool issuer = (wave == 0);

    v8f z = {0.f, 0.f, 0.f, 0.f, 0.f, 0.f, 0.f, 0.f};
    v8f acc[8];
#pragma unroll
    for (int i = 0; i < 8; ++i) acc[i] = z;

    if (issuer) {   // prologue: stage tiles for k0 = 0 into buffer 0
        tdm_load_2d(lds_addr(&At[0][0]), Ag, K, (unsigned)(M - blockRow), K, 32, 256);
        tdm_load_2d(lds_addr(&Bt[0][0]), Wg, K, (unsigned)(N - n0),       K, 32, 64);
    }

    for (int k0 = 0; k0 < K; k0 += 32) {
        int cur = (k0 >> 5) & 1;
        if (issuer) {
            if (k0 + 32 < K) {   // stage next tiles, then wait for current pair
                tdm_load_2d(lds_addr(&At[cur ^ 1][0]), Ag + k0 + 32, K, (unsigned)(M - blockRow), K, 32, 256);
                tdm_load_2d(lds_addr(&Bt[cur ^ 1][0]), Wg + k0 + 32, K, (unsigned)(N - n0),       K, 32, 64);
                __builtin_amdgcn_s_wait_tensorcnt(2);
            } else {
                __builtin_amdgcn_s_wait_tensorcnt(0);
            }
        }
        __syncthreads();         // current tiles resident for all waves

        // read 4 B frags from LDS: lane<16 col=lane K 0..7,16..23 ; lane>=16 K 8..15,24..31
        v16bf bfr[4];
#pragma unroll
        for (int t = 0; t < 4; ++t) {
            const bf16* p = &Bt[cur][(size_t)(t * 16 + (lane & 15)) * 32 + ((lane & 16) ? 8 : 0)];
            union { v16bf v; v8bf h[2]; } u;
            u.h[0] = *(const v8bf*)(p);
            u.h[1] = *(const v8bf*)(p + 16);
            bfr[t] = u.v;
        }
        // A frags from LDS (per-wave rows, tile stride 32)
        v16bf a0 = load_a_frag(&At[cur][(size_t)(wave * 32 +  0) * 32], 32, lane);
        v16bf a1 = load_a_frag(&At[cur][(size_t)(wave * 32 + 16) * 32], 32, lane);

#pragma unroll
        for (int t = 0; t < 4; ++t) {
            acc[t]     = __builtin_amdgcn_wmma_f32_16x16x32_bf16(false, a0, false, bfr[t], (short)0, acc[t],     false, false);
            acc[4 + t] = __builtin_amdgcn_wmma_f32_16x16x32_bf16(false, a1, false, bfr[t], (short)0, acc[4 + t], false, false);
        }
        wait_ds0();              // frag reads drained from LDS
        __syncthreads();         // safe to overwrite the other buffer next iter
    }

    // epilogue
    int hi8 = (lane >> 4) & 1;
#pragma unroll
    for (int g = 0; g < 2; ++g) {
        int rbase = wrow + g * 16 + hi8 * 8;
#pragma unroll
        for (int t = 0; t < 4; ++t) {
            int col = n0 + t * 16 + (lane & 15);
            float bs = bias[col];
#pragma unroll
            for (int r = 0; r < 8; ++r) {
                size_t o = (size_t)(rbase + r) * N + col;
                float vv = acc[g * 4 + t][r] + bs;
                if (MODE == 0) {
                    outf[o] = vv;
                } else if (MODE == 1) {
                    outb[o] = (bf16)vv;
                } else if (MODE == 2) {
                    outf[o] = res[o] + vv;
                } else {
                    float gg = 0.5f * vv * (1.0f + erff(vv * 0.70710678118654752f));
                    outb[o] = (bf16)gg;
                }
            }
        }
    }
}

// =====================================================================
// Flash attention, causal. One wave per (b, h, 16-row q tile).
// V tiles streamed into LDS by the Tensor Data Mover (OOB rows zero-fill).
// =====================================================================
__global__ void attn_kernel(const bf16* __restrict__ q,
                            const bf16* __restrict__ k,
                            const bf16* __restrict__ v,
                            bf16* __restrict__ y) {
    __shared__ __align__(16) bf16 Vt[32 * 64];   // V tile: 32 keys x 64 d
    __shared__ __align__(16) bf16 Pt[16 * 32];   // P tile: 16 q x 32 keys

    int lane = threadIdx.x;       // 32 threads
    int q0 = blockIdx.x * 16;
    int bh = blockIdx.y;
    int b = bh / HNUM, h = bh % HNUM;
    size_t tokbase = (size_t)b * TSEQ;
    int hcol = h * HDIM;

    // Q A-fragments for d-chunks 0..31 and 32..63
    v16bf aq0, aq1;
    {
        const bf16* qb = q + (tokbase + q0 + (lane & 15)) * EDIM + hcol;
        int koff = (lane & 16) ? 8 : 0;
        union { v16bf vv; v8bf h2[2]; } u;
        u.h2[0] = *(const v8bf*)(qb + 0 + koff);
        u.h2[1] = *(const v8bf*)(qb + 0 + koff + 16);
        aq0 = u.vv;
        u.h2[0] = *(const v8bf*)(qb + 32 + koff);
        u.h2[1] = *(const v8bf*)(qb + 32 + koff + 16);
        aq1 = u.vv;
    }

    float m[8], l[8];
#pragma unroll
    for (int r = 0; r < 8; ++r) { m[r] = -1e30f; l[r] = 0.f; }
    v8f z = {0.f, 0.f, 0.f, 0.f, 0.f, 0.f, 0.f, 0.f};
    v8f O0 = z, O1 = z, O2 = z, O3 = z;

    int hi8 = (lane >> 4) & 1;
    unsigned vt_base = lds_addr(&Vt[0]);
    const float scale = 0.125f;   // 1/sqrt(64)

    for (int j = 0; j <= q0 + 15; j += 32) {
        // ---- TDM: stage V tile (32 keys x 64 d) into LDS; rows >= T zero-fill
        tdm_load_2d(vt_base, v + (tokbase + j) * EDIM + hcol,
                    64, (unsigned)(TSEQ - j), EDIM, 64, 32);

        // ---- S = Q K^T for keys [j, j+32): two 16x16 tiles
        v8f s0 = z, s1 = z;
        {
            int key0 = j + (lane & 15);      if (key0 > TSEQ - 1) key0 = TSEQ - 1;
            int key1 = j + 16 + (lane & 15); if (key1 > TSEQ - 1) key1 = TSEQ - 1;
            int koff = (lane & 16) ? 8 : 0;
            const bf16* kb0 = k + (tokbase + key0) * EDIM + hcol;
            const bf16* kb1 = k + (tokbase + key1) * EDIM + hcol;
            union { v16bf vv; v8bf h2[2]; } u;
            u.h2[0] = *(const v8bf*)(kb0 + koff);       u.h2[1] = *(const v8bf*)(kb0 + koff + 16);
            s0 = __builtin_amdgcn_wmma_f32_16x16x32_bf16(false, aq0, false, u.vv, (short)0, s0, false, false);
            u.h2[0] = *(const v8bf*)(kb0 + 32 + koff);  u.h2[1] = *(const v8bf*)(kb0 + 32 + koff + 16);
            s0 = __builtin_amdgcn_wmma_f32_16x16x32_bf16(false, aq1, false, u.vv, (short)0, s0, false, false);
            u.h2[0] = *(const v8bf*)(kb1 + koff);       u.h2[1] = *(const v8bf*)(kb1 + koff + 16);
            s1 = __builtin_amdgcn_wmma_f32_16x16x32_bf16(false, aq0, false, u.vv, (short)0, s1, false, false);
            u.h2[0] = *(const v8bf*)(kb1 + 32 + koff);  u.h2[1] = *(const v8bf*)(kb1 + 32 + koff + 16);
            s1 = __builtin_amdgcn_wmma_f32_16x16x32_bf16(false, aq1, false, u.vv, (short)0, s1, false, false);
        }

        // ---- masked online softmax update (per q-row, DPP reductions)
        float alpha[8];
        int col = lane & 15;
#pragma unroll
        for (int r = 0; r < 8; ++r) {
            int qrow = q0 + hi8 * 8 + r;
            int key0 = j + col, key1 = j + 16 + col;
            float v0 = (key0 <= qrow) ? s0[r] * scale : -1e30f;
            float v1 = (key1 <= qrow) ? s1[r] * scale : -1e30f;
            float t = rmax16(fmaxf(v0, v1));
            float nm = fmaxf(m[r], t);
            alpha[r] = __expf(m[r] - nm);
            m[r] = nm;
            float p0 = __expf(v0 - nm);
            float p1 = __expf(v1 - nm);
            l[r] = l[r] * alpha[r] + rsum16(p0 + p1);
            Pt[(hi8 * 8 + r) * 32 + col]      = (bf16)p0;
            Pt[(hi8 * 8 + r) * 32 + col + 16] = (bf16)p1;
        }

        // ---- rescale O accumulators register-wise (reg r == q-row r)
#pragma unroll
        for (int r = 0; r < 8; ++r) {
            O0[r] *= alpha[r]; O1[r] *= alpha[r]; O2[r] *= alpha[r]; O3[r] *= alpha[r];
        }

        // ---- A-frag of P from LDS
        v16bf ap;
        {
            const bf16* p = Pt + (lane & 15) * 32 + ((lane & 16) ? 8 : 0);
            union { v16bf vv; v8bf h2[2]; } u;
            u.h2[0] = *(const v8bf*)(p);
            u.h2[1] = *(const v8bf*)(p + 16);
            ap = u.vv;
        }

        __builtin_amdgcn_s_wait_tensorcnt(0);   // V tile resident

        // ---- O += P @ V, four 16-wide d tiles; B-frags gathered from LDS
        int kk0 = (lane & 16) ? 8 : 0;
#pragma unroll
        for (int t = 0; t < 4; ++t) {
            int dcol = t * 16 + (lane & 15);
            union { v16bf vv; bf16 e[16]; } bv;
#pragma unroll
            for (int i = 0; i < 16; ++i) bv.e[i] = Vt[(kk0 + i + (i >= 8 ? 8 : 0)) * 64 + dcol];
            v8f* Op = (t == 0) ? &O0 : (t == 1) ? &O1 : (t == 2) ? &O2 : &O3;
            *Op = __builtin_amdgcn_wmma_f32_16x16x32_bf16(false, ap, false, bv.vv, (short)0, *Op, false, false);
        }
    }

    // ---- normalize and store y (bf16)
    float inv[8];
#pragma unroll
    for (int r = 0; r < 8; ++r) inv[r] = 1.0f / l[r];
    v8f Os[4] = {O0, O1, O2, O3};
#pragma unroll
    for (int t = 0; t < 4; ++t) {
        int dcol = hcol + t * 16 + (lane & 15);
#pragma unroll
        for (int r = 0; r < 8; ++r) {
            int tokrow = q0 + hi8 * 8 + r;
            y[(tokbase + tokrow) * EDIM + dcol] = (bf16)(Os[t][r] * inv[r]);
        }
    }
}

// =====================================================================
// host side
// =====================================================================
extern "C" void kernel_launch(void* const* d_in, const int* in_sizes, int n_in,
                              void* d_out, int out_size, void* d_ws, size_t ws_size,
                              hipStream_t stream) {
    const int*   idx   = (const int*)  d_in[0];
    const float* tok   = (const float*)d_in[1];
    const float* pos   = (const float*)d_in[2];
    const float* ln1g  = (const float*)d_in[3];
    const float* ln1b  = (const float*)d_in[4];
    const float* Wq    = (const float*)d_in[5];
    const float* bq    = (const float*)d_in[6];
    const float* Wk    = (const float*)d_in[7];
    const float* bk    = (const float*)d_in[8];
    const float* Wv    = (const float*)d_in[9];
    const float* bv    = (const float*)d_in[10];
    const float* Wp    = (const float*)d_in[11];
    const float* bp    = (const float*)d_in[12];
    const float* ln2g  = (const float*)d_in[13];
    const float* ln2b  = (const float*)d_in[14];
    const float* W1    = (const float*)d_in[15];
    const float* b1    = (const float*)d_in[16];
    const float* W2    = (const float*)d_in[17];
    const float* b2    = (const float*)d_in[18];
    const float* lnfg  = (const float*)d_in[19];
    const float* lnfb  = (const float*)d_in[20];
    const float* lmb   = (const float*)d_in[21];
    float* logits = (float*)d_out;

    // ---- workspace carve-up
    char* ws = (char*)d_ws;
    size_t off = 0;
    auto take = [&](size_t bytes) -> void* {
        void* p = ws + off;
        off += (bytes + 255) & ~(size_t)255;
        return p;
    };
    const size_t EE = (size_t)EDIM * EDIM;               // 589824
    bf16* tokbf = (bf16*)take((size_t)VCAB * EDIM * 2);
    bf16* wqbf  = (bf16*)take(LNUM * EE * 2);
    bf16* wkbf  = (bf16*)take(LNUM * EE * 2);
    bf16* wvbf  = (bf16*)take(LNUM * EE * 2);
    bf16* wpbf  = (bf16*)take(LNUM * EE * 2);
    bf16* w1bf  = (bf16*)take(LNUM * 4 * EE * 2);
    bf16* w2bf  = (bf16*)take(LNUM * 4 * EE * 2);
    float* x    = (float*)take((size_t)MTOK * EDIM * 4);
    bf16* hbf   = (bf16*)take((size_t)MTOK * EDIM * 2);
    bf16* qbf   = (bf16*)take((size_t)MTOK * EDIM * 2);
    bf16* kbf   = (bf16*)take((size_t)MTOK * EDIM * 2);
    bf16* vbf   = (bf16*)take((size_t)MTOK * EDIM * 2);
    bf16* ybf   = (bf16*)take((size_t)MTOK * EDIM * 2);
    bf16* hidbf = (bf16*)take((size_t)MTOK * 4 * EDIM * 2);

    // ---- 1) convert all weights fp32 -> bf16
    f32_to_bf16_kernel<<<4096, 256, 0, stream>>>(tok, tokbf, VCAB * EDIM);
    f32_to_bf16_kernel<<<4096, 256, 0, stream>>>(Wq, wqbf, (int)(LNUM * EE));
    f32_to_bf16_kernel<<<4096, 256, 0, stream>>>(Wk, wkbf, (int)(LNUM * EE));
    f32_to_bf16_kernel<<<4096, 256, 0, stream>>>(Wv, wvbf, (int)(LNUM * EE));
    f32_to_bf16_kernel<<<4096, 256, 0, stream>>>(Wp, wpbf, (int)(LNUM * EE));
    f32_to_bf16_kernel<<<4096, 256, 0, stream>>>(W1, w1bf, (int)(LNUM * 4 * EE));
    f32_to_bf16_kernel<<<4096, 256, 0, stream>>>(W2, w2bf, (int)(LNUM * 4 * EE));

    // ---- 2) embedding
    embed_kernel<<<MTOK, 256, 0, stream>>>(idx, tok, pos, x);

    // ---- 3) transformer layers
    dim3 gProj(EDIM / 64, MTOK / 256);       // 12 x 16
    dim3 gMlp1(4 * EDIM / 64, MTOK / 256);   // 48 x 16
    dim3 gAttn(TSEQ / 16, BBAT * HNUM);      // 64 x 48
    for (int l = 0; l < LNUM; ++l) {
        const bf16* wq_l = wqbf + l * EE;
        const bf16* wk_l = wkbf + l * EE;
        const bf16* wv_l = wvbf + l * EE;
        const bf16* wp_l = wpbf + l * EE;
        const bf16* w1_l = w1bf + l * 4 * EE;
        const bf16* w2_l = w2bf + l * 4 * EE;

        layernorm_bf16_kernel<<<MTOK, 256, 0, stream>>>(x, ln1g + l * EDIM, ln1b + l * EDIM, hbf);

        gemm_wmma_kernel<1><<<gProj, 256, 0, stream>>>(hbf, wq_l, bq + l * EDIM, nullptr, nullptr, qbf, MTOK, EDIM, EDIM);
        gemm_wmma_kernel<1><<<gProj, 256, 0, stream>>>(hbf, wk_l, bk + l * EDIM, nullptr, nullptr, kbf, MTOK, EDIM, EDIM);
        gemm_wmma_kernel<1><<<gProj, 256, 0, stream>>>(hbf, wv_l, bv + l * EDIM, nullptr, nullptr, vbf, MTOK, EDIM, EDIM);

        attn_kernel<<<gAttn, 32, 0, stream>>>(qbf, kbf, vbf, ybf);

        gemm_wmma_kernel<2><<<gProj, 256, 0, stream>>>(ybf, wp_l, bp + l * EDIM, x, x, nullptr, MTOK, EDIM, EDIM);

        layernorm_bf16_kernel<<<MTOK, 256, 0, stream>>>(x, ln2g + l * EDIM, ln2b + l * EDIM, hbf);

        gemm_wmma_kernel<3><<<gMlp1, 256, 0, stream>>>(hbf, w1_l, b1 + l * 4 * EDIM, nullptr, nullptr, hidbf, MTOK, 4 * EDIM, EDIM);
        gemm_wmma_kernel<2><<<gProj, 256, 0, stream>>>(hidbf, w2_l, b2 + l * EDIM, x, x, nullptr, MTOK, EDIM, 4 * EDIM);
    }

    // ---- 4) final LN + tied LM head
    layernorm_bf16_kernel<<<MTOK, 256, 0, stream>>>(x, lnfg, lnfb, hbf);
    dim3 gHead(VCAB / 64, MTOK / 256);       // 500 x 16
    gemm_wmma_kernel<0><<<gHead, 256, 0, stream>>>(hbf, tokbf, lmb, nullptr, logits, nullptr, MTOK, VCAB, EDIM);
}
